// DynamicModel_79955111182517
// MI455X (gfx1250) — matrix operand order repaired and checked by reading, not compiled, not run on hardware
//
#include <hip/hip_runtime.h>
#include <hip/hip_bf16.h>

typedef __bf16 bf16;
typedef __attribute__((ext_vector_type(16))) __bf16 v16bf;
typedef __attribute__((ext_vector_type(8)))  __bf16 v8bf;
typedef __attribute__((ext_vector_type(8)))  float  v8f;

#define WMMA_BF16(a, b, c) \
  __builtin_amdgcn_wmma_f32_16x16x32_bf16(false, (a), false, (b), (short)0, (c), false, false)

static constexpr int B_  = 8;
static constexpr int P_  = 512;
static constexpr int KNN = 16;

// ---------------------------------------------------------------------------
// A-fragment layout (16x16x32 bf16 A, wave32), per ISA 7.12.2:
//   lanes 0-15 : M=lane,  elem j -> K = j (j<8) / j+8 (j>=8)
//   lanes 16-31: M=l-16,  elem j -> K = j+8 (j<8) / j+16 (j>=8)
// We store A tiles in LDS so that lane l reads elements [l*16 .. l*16+15]
// contiguously. afrag_index maps a logical (row, k) to that layout.
// ---------------------------------------------------------------------------
__device__ __forceinline__ int afrag_index(int row, int k) {
  int t  = k >> 5;          // 32-wide K tile
  int kk = k & 31;
  int hi = (kk >> 3) & 1;                                // lane-half
  int j  = (kk & 7) | (((kk >> 4) & 1) << 3);            // element slot
  return (t << 9) + ((row + (hi << 4)) << 4) + j;
}

// Apply bias + ReLU + BN(scale,shift) to a C tile and scatter it into the
// next layer's A-fragment LDS buffer as bf16.
__device__ __forceinline__ void store_bn_relu_afrag(
    bf16* dst, const v8f acc, int nt, int lane,
    const float* __restrict__ bias, const float* __restrict__ g,
    const float* __restrict__ bt) {
  int col = (nt << 4) + (lane & 15);
  float bb = bias[col], gg = g[col], tt = bt[col];
  int hiM = lane >> 4;
  int t   = col >> 5;
  int kk  = col & 31;
  int hi2 = (kk >> 3) & 1;
  int j2  = (kk & 7) | (((kk >> 4) & 1) << 3);
#pragma unroll
  for (int r = 0; r < 8; ++r) {
    float v = fmaxf(acc[r] + bb, 0.f) * gg + tt;
    int M = r + (hiM << 3);
    dst[(t << 9) + ((M + (hi2 << 4)) << 4) + j2] = (bf16)v;
  }
}

// ---------------------------------------------------------------------------
// Conversion / weight pre-swizzle kernels (one-shot, bandwidth trivial)
// ---------------------------------------------------------------------------
__global__ void k_f32_to_bf16(const float* __restrict__ in, bf16* __restrict__ out, int n) {
  int i = blockIdx.x * blockDim.x + threadIdx.x;
  if (i < n) out[i] = (bf16)in[i];
}

// B-fragment layout: lane l holds column N = nbase + (l&15), elems j are
// K = kbase + 8*(l>>4) + (j&7) + 16*(j>>3).  Stored so a fragment is one 32B load:
// out[((nt*KT + kt) * 512) + lane*16 + j]
__global__ void k_swz_weights(const float* __restrict__ W, bf16* __restrict__ out,
                              int Kdim, int N) {
  int i = blockIdx.x * blockDim.x + threadIdx.x;
  if (i >= Kdim * N) return;
  int j    = i & 15;
  int lane = (i >> 4) & 31;
  int tile = i >> 9;
  int KT   = Kdim >> 5;
  int kt   = tile % KT;
  int nt   = tile / KT;
  int hi   = lane >> 4;
  int k    = (kt << 5) + (hi << 3) + (j & 7) + ((j >> 3) << 4);
  int n    = (nt << 4) + (lane & 15);
  out[i] = (bf16)W[(size_t)k * N + n];
}

__global__ void k_norms(const bf16* __restrict__ x, float* __restrict__ nrm,
                        int npts, int C) {
  int i = blockIdx.x * blockDim.x + threadIdx.x;
  if (i >= npts) return;
  const bf16* r = x + (size_t)i * C;
  float s = 0.f;
  for (int k = 0; k < C; ++k) { float v = (float)r[k]; s += v * v; }
  nrm[i] = s;
}

// ---------------------------------------------------------------------------
// Pairwise squared distances via WMMA Gram matrix.
// grid = B*32*32 blocks (one wave each); block (b, it, jt) computes a 16x16
// tile of d2[b] = |xi|^2 + |xj|^2 - 2 * x x^T.
// ---------------------------------------------------------------------------
__global__ __launch_bounds__(32) void k_pairdist(
    const bf16* __restrict__ x, const float* __restrict__ nrm,
    float* __restrict__ d2, int C) {
  int blk  = blockIdx.x;
  int jt   = blk & 31;
  int it   = (blk >> 5) & 31;
  int b    = blk >> 10;
  int lane = threadIdx.x & 31;
  int hi   = lane >> 4;
  int ibase = b * P_ + (it << 4);
  int jbase = b * P_ + (jt << 4);
  const bf16* arow = x + (size_t)(ibase + (lane & 15)) * C;
  const bf16* brow = x + (size_t)(jbase + (lane & 15)) * C;
  v8f acc = {};
  for (int k0 = 0; k0 < C; k0 += 32) {
    v8bf a0 = *(const v8bf*)(arow + k0 + 16 * hi);
    v8bf a1 = *(const v8bf*)(arow + k0 + 16 * hi + 8);
    v16bf af;
#pragma unroll
    for (int j = 0; j < 8; ++j) { af[j] = a0[j]; af[j + 8] = a1[j]; }
    v16bf bfv = *(const v16bf*)(brow + k0 + (hi << 4));
    acc = WMMA_BF16(af, bfv, acc);
  }
  int jl = (jt << 4) + (lane & 15);
  float nj = nrm[jbase + (lane & 15)];
#pragma unroll
  for (int r = 0; r < 8; ++r) {
    int i = ibase + r + (hi << 3);
    d2[(size_t)i * P_ + jl] = nrm[i] + nj - 2.f * acc[r];
  }
}

// ---------------------------------------------------------------------------
// Top-K (k=16) smallest per row; fully register-resident insertion sort.
// ---------------------------------------------------------------------------
__global__ void k_topk(const float* __restrict__ d2, int* __restrict__ idx) {
  int row = blockIdx.x * blockDim.x + threadIdx.x;
  if (row >= B_ * P_) return;
  const float* dr = d2 + (size_t)row * P_;
  float bd[KNN]; int bi[KNN];
#pragma unroll
  for (int t = 0; t < KNN; ++t) { bd[t] = 3.0e38f; bi[t] = 0; }
  for (int j = 0; j < P_; ++j) {
    float d = dr[j];
    if (d < bd[KNN - 1]) {
      bd[KNN - 1] = d; bi[KNN - 1] = j;
#pragma unroll
      for (int t = KNN - 1; t > 0; --t) {
        if (bd[t] < bd[t - 1]) {
          float td = bd[t]; bd[t] = bd[t - 1]; bd[t - 1] = td;
          int   ti = bi[t]; bi[t] = bi[t - 1]; bi[t - 1] = ti;
        }
      }
    }
  }
#pragma unroll
  for (int t = 0; t < KNN; ++t) idx[row * KNN + t] = bi[t];
}

// ---------------------------------------------------------------------------
// EdgeConv1: per point, A = 16 neighbor messages [xi, xj-xi] (16x2048),
// W 2048x512 -> 16x512, ReLU+BN, max over the 16 rows. 1 workgroup / point.
// ---------------------------------------------------------------------------
__global__ __launch_bounds__(256) void k_edgeconv1(
    const bf16* __restrict__ xb, const int* __restrict__ idx,
    const bf16* __restrict__ wsz, const float* __restrict__ bias,
    const float* __restrict__ g, const float* __restrict__ bt,
    bf16* __restrict__ out) {
  __shared__ alignas(32) bf16 ldsA[16 * 2048];   // 64 KB, fragment layout
  int gp  = blockIdx.x;
  int b   = gp >> 9;
  int tid = threadIdx.x;
  const bf16* xi = xb + (size_t)gp * 1024;
  for (int e = tid; e < 16 * 2048; e += 256) {
    int row = e >> 11, k = e & 2047;
    int jn = idx[gp * KNN + row];
    const bf16* xj = xb + (size_t)((b << 9) + jn) * 1024;
    float v = (k < 1024) ? (float)xi[k]
                         : ((float)xj[k - 1024] - (float)xi[k - 1024]);
    ldsA[afrag_index(row, k)] = (bf16)v;
  }
  __syncthreads();
  int wave = tid >> 5, lane = tid & 31;
  for (int nt = wave; nt < 32; nt += 8) {
    v8f acc = {};
    const bf16* wp = wsz + (((size_t)nt * 64) << 9) + lane * 16;
#pragma unroll 4
    for (int kt = 0; kt < 64; ++kt) {
      v16bf a = *(const v16bf*)&ldsA[(kt << 9) + lane * 16];
      v16bf w = *(const v16bf*)(wp + ((size_t)kt << 9));
      acc = WMMA_BF16(a, w, acc);
    }
    int col = (nt << 4) + (lane & 15);
    float bb = bias[col], gg = g[col], tt = bt[col];
    float m = -3.0e38f;
#pragma unroll
    for (int r = 0; r < 8; ++r)
      m = fmaxf(m, fmaxf(acc[r] + bb, 0.f) * gg + tt);
    m = fmaxf(m, __shfl_xor(m, 16, 32));       // merge M 0-7 with 8-15
    if (lane < 16) out[(size_t)gp * 512 + col] = (bf16)m;
  }
}

// ---------------------------------------------------------------------------
// EdgeConv2: 3 chained blocks 1024->512->1024->1024 inside one workgroup,
// LDS ping-pong (reg0+reg1 48KB total), max over 16 rows at the end.
// ---------------------------------------------------------------------------
__global__ __launch_bounds__(256) void k_edgeconv2(
    const bf16* __restrict__ x1, const int* __restrict__ idx,
    const bf16* __restrict__ w1, const float* __restrict__ b1,
    const float* __restrict__ g1, const float* __restrict__ t1,
    const bf16* __restrict__ w2, const float* __restrict__ b2,
    const float* __restrict__ g2, const float* __restrict__ t2,
    const bf16* __restrict__ w3, const float* __restrict__ b3,
    const float* __restrict__ g3, const float* __restrict__ t3,
    bf16* __restrict__ out) {
  __shared__ alignas(32) bf16 reg0[16 * 1024];   // A0, reused for h2
  __shared__ alignas(32) bf16 reg1[16 * 512];    // h1
  int gp  = blockIdx.x;
  int b   = gp >> 9;
  int tid = threadIdx.x;
  const bf16* xi = x1 + (size_t)gp * 512;
  for (int e = tid; e < 16 * 1024; e += 256) {
    int row = e >> 10, k = e & 1023;
    int jn = idx[gp * KNN + row];
    const bf16* xj = x1 + (size_t)((b << 9) + jn) * 512;
    float v = (k < 512) ? (float)xi[k]
                        : ((float)xj[k - 512] - (float)xi[k - 512]);
    reg0[afrag_index(row, k)] = (bf16)v;
  }
  __syncthreads();
  int wave = tid >> 5, lane = tid & 31;
  // layer 1: 1024 -> 512
  for (int nt = wave; nt < 32; nt += 8) {
    v8f acc = {};
    for (int kt = 0; kt < 32; ++kt) {
      v16bf a = *(const v16bf*)&reg0[(kt << 9) + lane * 16];
      v16bf w = *(const v16bf*)&w1[(((size_t)nt * 32 + kt) << 9) + lane * 16];
      acc = WMMA_BF16(a, w, acc);
    }
    store_bn_relu_afrag(reg1, acc, nt, lane, b1, g1, t1);
  }
  __syncthreads();
  // layer 2: 512 -> 1024 (writes over A0 region)
  for (int nt = wave; nt < 64; nt += 8) {
    v8f acc = {};
    for (int kt = 0; kt < 16; ++kt) {
      v16bf a = *(const v16bf*)&reg1[(kt << 9) + lane * 16];
      v16bf w = *(const v16bf*)&w2[(((size_t)nt * 16 + kt) << 9) + lane * 16];
      acc = WMMA_BF16(a, w, acc);
    }
    store_bn_relu_afrag(reg0, acc, nt, lane, b2, g2, t2);
  }
  __syncthreads();
  // layer 3: 1024 -> 1024, then max over the 16 neighbor rows
  for (int nt = wave; nt < 64; nt += 8) {
    v8f acc = {};
    for (int kt = 0; kt < 32; ++kt) {
      v16bf a = *(const v16bf*)&reg0[(kt << 9) + lane * 16];
      v16bf w = *(const v16bf*)&w3[(((size_t)nt * 32 + kt) << 9) + lane * 16];
      acc = WMMA_BF16(a, w, acc);
    }
    int col = (nt << 4) + (lane & 15);
    float bb = b3[col], gg = g3[col], tt = t3[col];
    float m = -3.0e38f;
#pragma unroll
    for (int r = 0; r < 8; ++r)
      m = fmaxf(m, fmaxf(acc[r] + bb, 0.f) * gg + tt);
    m = fmaxf(m, __shfl_xor(m, 16, 32));
    if (lane < 16) out[(size_t)gp * 1024 + col] = (bf16)m;
  }
}

// ---------------------------------------------------------------------------
// lin1 (concat[x1,x2] 1536 -> 2048) fused with global max pool over P=512.
// grid = B * 128 column tiles; 8 waves each own 4 row tiles; cross-wave max
// via LDS; output pooled[B,2048] fp32.
// ---------------------------------------------------------------------------
__global__ __launch_bounds__(256) void k_lin1_pool(
    const bf16* __restrict__ x1, const bf16* __restrict__ x2,
    const bf16* __restrict__ wsz, const float* __restrict__ bias,
    const float* __restrict__ g, const float* __restrict__ bt,
    float* __restrict__ pooled) {
  __shared__ float red[8 * 16];
  int b    = blockIdx.x >> 7;
  int nt   = blockIdx.x & 127;
  int tid  = threadIdx.x;
  int wave = tid >> 5, lane = tid & 31, hi = lane >> 4;
  int col  = (nt << 4) + (lane & 15);
  float bb = bias[col], gg = g[col], tt = bt[col];
  float m  = -3.0e38f;
  for (int rt = wave; rt < 32; rt += 8) {
    int p = (b << 9) + (rt << 4) + (lane & 15);
    const bf16* r1 = x1 + (size_t)p * 512;
    const bf16* r2 = x2 + (size_t)p * 1024;
    v8f acc = {};
    for (int kt = 0; kt < 48; ++kt) {
      int k0 = kt << 5;
      const bf16* src; int off;
      if (k0 < 512) { src = r1; off = k0; } else { src = r2; off = k0 - 512; }
      v8bf a0 = *(const v8bf*)(src + off + 8 * hi);
      v8bf a1 = *(const v8bf*)(src + off + 16 + 8 * hi);
      v16bf a;
#pragma unroll
      for (int j = 0; j < 8; ++j) { a[j] = a0[j]; a[j + 8] = a1[j]; }
      v16bf w = *(const v16bf*)&wsz[(((size_t)nt * 48 + kt) << 9) + lane * 16];
      acc = WMMA_BF16(a, w, acc);
    }
#pragma unroll
    for (int r = 0; r < 8; ++r)
      m = fmaxf(m, fmaxf(acc[r] + bb, 0.f) * gg + tt);
  }
  m = fmaxf(m, __shfl_xor(m, 16, 32));
  if (lane < 16) red[wave * 16 + lane] = m;
  __syncthreads();
  if (tid < 16) {
    float v = red[tid];
#pragma unroll
    for (int w2 = 1; w2 < 8; ++w2) v = fmaxf(v, red[w2 * 16 + tid]);
    pooled[(size_t)b * 2048 + (nt << 4) + tid] = v;
  }
}

// ---------------------------------------------------------------------------
// Head MLPs (8 rows — negligible FLOPs, stay fp32 for output precision)
// ---------------------------------------------------------------------------
__global__ void k_mlp_head(const float* __restrict__ in, const float* __restrict__ W,
                           const float* __restrict__ bias, const float* __restrict__ g,
                           const float* __restrict__ bt, float* __restrict__ out,
                           int cin, int cout) {
  int o = blockIdx.x * blockDim.x + threadIdx.x;
  if (o >= B_ * cout) return;
  int b = o / cout, c = o - b * cout;
  const float* r = in + (size_t)b * cin;
  float s = 0.f;
  for (int k = 0; k < cin; ++k) s += r[k] * W[(size_t)k * cout + c];
  out[o] = fmaxf(s + bias[c], 0.f) * g[c] + bt[c];
}

__global__ void k_final(const float* __restrict__ in, const float* __restrict__ W,
                        const float* __restrict__ bias, float* __restrict__ out) {
  int o = blockIdx.x * blockDim.x + threadIdx.x;
  if (o >= B_ * 10) return;
  int b = o / 10, c = o - b * 10;
  const float* r = in + b * 512;
  float s = 0.f;
  for (int k = 0; k < 512; ++k) s += r[k] * W[k * 10 + c];
  out[o] = s + bias[c];
}

// ---------------------------------------------------------------------------
extern "C" void kernel_launch(void* const* d_in, const int* in_sizes, int n_in,
                              void* d_out, int out_size, void* d_ws, size_t ws_size,
                              hipStream_t stream) {
  (void)in_sizes; (void)n_in; (void)out_size; (void)ws_size;
  // Input order: x, batch, num_graphs, conv1{W,b,g,bt}, conv2{3 blocks},
  // lin1{...}, head1{...}, head2{...}, headW, headb
  const float* x    = (const float*)d_in[0];
  const float* W1   = (const float*)d_in[3];
  const float* b1   = (const float*)d_in[4];
  const float* g1   = (const float*)d_in[5];
  const float* t1   = (const float*)d_in[6];
  const float* W2a  = (const float*)d_in[7];
  const float* b2a  = (const float*)d_in[8];
  const float* g2a  = (const float*)d_in[9];
  const float* t2a  = (const float*)d_in[10];
  const float* W2b  = (const float*)d_in[11];
  const float* b2b  = (const float*)d_in[12];
  const float* g2b  = (const float*)d_in[13];
  const float* t2b  = (const float*)d_in[14];
  const float* W2c  = (const float*)d_in[15];
  const float* b2c  = (const float*)d_in[16];
  const float* g2c  = (const float*)d_in[17];
  const float* t2c  = (const float*)d_in[18];
  const float* Wl   = (const float*)d_in[19];
  const float* bl   = (const float*)d_in[20];
  const float* gl   = (const float*)d_in[21];
  const float* tl   = (const float*)d_in[22];
  const float* Wh1  = (const float*)d_in[23];
  const float* bh1  = (const float*)d_in[24];
  const float* gh1  = (const float*)d_in[25];
  const float* th1  = (const float*)d_in[26];
  const float* Wh2  = (const float*)d_in[27];
  const float* bh2  = (const float*)d_in[28];
  const float* gh2  = (const float*)d_in[29];
  const float* th2  = (const float*)d_in[30];
  const float* WH   = (const float*)d_in[31];
  const float* bH   = (const float*)d_in[32];

  char* base = (char*)d_ws;
  size_t off = 0;
  auto alloc = [&](size_t bytes) -> void* {
    void* p = base + off;
    off += (bytes + 255) & ~(size_t)255;
    return p;
  };
  bf16*  xb     = (bf16*)alloc((size_t)4096 * 1024 * 2);
  bf16*  w1s    = (bf16*)alloc((size_t)2048 * 512 * 2);
  bf16*  w2as   = (bf16*)alloc((size_t)1024 * 512 * 2);
  bf16*  w2bs   = (bf16*)alloc((size_t)512 * 1024 * 2);
  bf16*  w2cs   = (bf16*)alloc((size_t)1024 * 1024 * 2);
  bf16*  wls    = (bf16*)alloc((size_t)1536 * 2048 * 2);
  float* d2     = (float*)alloc((size_t)4096 * 512 * 4);
  float* nrm    = (float*)alloc((size_t)4096 * 4);
  int*   idx    = (int*)alloc((size_t)4096 * 16 * 4);
  bf16*  x1b    = (bf16*)alloc((size_t)4096 * 512 * 2);
  bf16*  x2b    = (bf16*)alloc((size_t)4096 * 1024 * 2);
  float* pooled = (float*)alloc((size_t)8 * 2048 * 4);
  float* h1     = (float*)alloc((size_t)8 * 1024 * 4);
  float* h2     = (float*)alloc((size_t)8 * 512 * 4);

  // stage 0: conversions + weight pre-swizzle (B-fragment layout)
  k_f32_to_bf16<<<(4096 * 1024 + 255) / 256, 256, 0, stream>>>(x, xb, 4096 * 1024);
  k_swz_weights<<<(2048 * 512 + 255) / 256, 256, 0, stream>>>(W1, w1s, 2048, 512);
  k_swz_weights<<<(1024 * 512 + 255) / 256, 256, 0, stream>>>(W2a, w2as, 1024, 512);
  k_swz_weights<<<(512 * 1024 + 255) / 256, 256, 0, stream>>>(W2b, w2bs, 512, 1024);
  k_swz_weights<<<(1024 * 1024 + 255) / 256, 256, 0, stream>>>(W2c, w2cs, 1024, 1024);
  k_swz_weights<<<(1536 * 2048 + 255) / 256, 256, 0, stream>>>(Wl, wls, 1536, 2048);

  // stage 1: EdgeConv1
  k_norms<<<16, 256, 0, stream>>>(xb, nrm, 4096, 1024);
  k_pairdist<<<8192, 32, 0, stream>>>(xb, nrm, d2, 1024);
  k_topk<<<16, 256, 0, stream>>>(d2, idx);
  k_edgeconv1<<<4096, 256, 0, stream>>>(xb, idx, w1s, b1, g1, t1, x1b);

  // stage 2: EdgeConv2
  k_norms<<<16, 256, 0, stream>>>(x1b, nrm, 4096, 512);
  k_pairdist<<<8192, 32, 0, stream>>>(x1b, nrm, d2, 512);
  k_topk<<<16, 256, 0, stream>>>(d2, idx);
  k_edgeconv2<<<4096, 256, 0, stream>>>(x1b, idx,
                                        w2as, b2a, g2a, t2a,
                                        w2bs, b2b, g2b, t2b,
                                        w2cs, b2c, g2c, t2c, x2b);

  // stage 3: lin1 + global max pool (fused)
  k_lin1_pool<<<1024, 256, 0, stream>>>(x1b, x2b, wls, bl, gl, tl, pooled);

  // stage 4: heads (fp32)
  k_mlp_head<<<(8 * 1024 + 255) / 256, 256, 0, stream>>>(pooled, Wh1, bh1, gh1, th1, h1, 2048, 1024);
  k_mlp_head<<<(8 * 512 + 255) / 256, 256, 0, stream>>>(h1, Wh2, bh2, gh2, th2, h2, 1024, 512);
  k_final<<<1, 128, 0, stream>>>(h2, WH, bH, (float*)d_out);
}
